// MambaBlock_19267223290332
// MI455X (gfx1250) — matrix operand order repaired and checked by reading, not compiled, not run on hardware
//
#include <hip/hip_runtime.h>
#include <hip/hip_bf16.h>

// ---------------------------------------------------------------------------
// Mamba-2 block forward for gfx1250 (MI455X), bf16 WMMA everywhere GEMM-shaped.
// ---------------------------------------------------------------------------

typedef __bf16 bf16;
typedef __attribute__((ext_vector_type(16))) __bf16 v16bf;
typedef __attribute__((ext_vector_type(8)))  float  v8f;

#define B_N   4
#define L_N   2048
#define D_MODEL 672
#define D_INNER 1344
#define D_STATE 128
#define NHEADS  24
#define HEADDIM 56
#define CHUNK   32
#define NCHUNK  64            // L_N / CHUNK
#define D_IN_PROJ 2968        // 2*D_INNER + 2*D_STATE + NHEADS
#define CONV_DIM 1600         // D_INNER + 2*D_STATE
#define MROWS 8192            // B_N * L_N
#define EPSF 1e-5f

// ---- bf16 <-> f32 via bit ops (no reliance on __bf16 arithmetic) -----------
static __device__ __forceinline__ bf16 f2bf(float f) {
    unsigned u = __builtin_bit_cast(unsigned, f);
    unsigned r = (u + 0x7FFFu + ((u >> 16) & 1u)) >> 16;
    unsigned short s = (unsigned short)r;
    return __builtin_bit_cast(bf16, s);
}
static __device__ __forceinline__ float bf2f(bf16 b) {
    unsigned short s = __builtin_bit_cast(unsigned short, b);
    unsigned u = ((unsigned)s) << 16;
    return __builtin_bit_cast(float, u);
}

// ---- WMMA fragment helpers (layouts per CDNA5 ISA 7.12.2) ------------------
// A (16x32 bf16, row-major src [.,ld]): lane r=lane&15 row, khalf=lane>>4.
static __device__ __forceinline__ v16bf frag_a(const bf16* A, int row0, int k0,
                                               int ld, int lane) {
    int r = lane & 15, hh = lane >> 4;
    const bf16* p = A + (size_t)(row0 + r) * ld + k0 + hh * 8;
    v16bf a;
#pragma unroll
    for (int j = 0; j < 8; ++j) { a[j] = p[j]; a[j + 8] = p[j + 16]; }
    return a;
}
// B (32x16 bf16) built from an [N,K]-row-major matrix W: Bmat[k][n] = W[n][k].
static __device__ __forceinline__ v16bf frag_b(const bf16* W, int n0, int k0,
                                               int ld, int lane) {
    int c = lane & 15, hh = lane >> 4;
    const bf16* p = W + (size_t)(n0 + c) * ld + k0 + hh * 8;
    v16bf b;
#pragma unroll
    for (int j = 0; j < 8; ++j) { b[j] = p[j]; b[j + 8] = p[j + 16]; }
    return b;
}
static __device__ __forceinline__ v16bf frag_b_clamp(const bf16* W, int n0, int k0,
                                                     int ld, int N, int lane) {
    int c = lane & 15, hh = lane >> 4;
    int n = n0 + c; n = (n < N) ? n : (N - 1);
    const bf16* p = W + (size_t)n * ld + k0 + hh * 8;
    v16bf b;
#pragma unroll
    for (int j = 0; j < 8; ++j) { b[j] = p[j]; b[j + 8] = p[j + 16]; }
    return b;
}
static __device__ __forceinline__ v8f wmma_bf16(v16bf a, v16bf b, v8f c) {
    return __builtin_amdgcn_wmma_f32_16x16x32_bf16(false, a, false, b,
                                                   (short)0, c, false, false);
}

// ---------------------------------------------------------------------------
// K0: convert projection weights to bf16
__global__ __launch_bounds__(256) void k_cvt(const float* __restrict__ w1, int n1,
                                             const float* __restrict__ w2, int n2,
                                             bf16* __restrict__ o1,
                                             bf16* __restrict__ o2) {
    int i = blockIdx.x * 256 + threadIdx.x;
    if (i < n1) o1[i] = f2bf(w1[i]);
    else if (i < n1 + n2) o2[i - n1] = f2bf(w2[i - n1]);
}

// K1: input RMSNorm -> bf16 activations
__global__ __launch_bounds__(256) void k_rmsnorm(const float* __restrict__ hs,
                                                 const float* __restrict__ w,
                                                 bf16* __restrict__ xo) {
    int row = blockIdx.x, tid = threadIdx.x;
    const float* x = hs + (size_t)row * D_MODEL;
    float s = 0.f;
    for (int i = tid; i < D_MODEL; i += 256) { float v = x[i]; s += v * v; }
    __shared__ float red[256];
    red[tid] = s; __syncthreads();
    for (int st = 128; st > 0; st >>= 1) {
        if (tid < st) red[tid] += red[tid + st];
        __syncthreads();
    }
    float rms = rsqrtf(red[0] / (float)D_MODEL + EPSF);
    for (int i = tid; i < D_MODEL; i += 256)
        xo[(size_t)row * D_MODEL + i] = f2bf(x[i] * rms * w[i]);
}

// K2/K10: bf16 WMMA GEMM  Out[M,N] = A[M,K] * W[N,K]^T (+ residual)
__global__ __launch_bounds__(256) void k_gemm_bf16(const bf16* __restrict__ A,
                                                   const bf16* __restrict__ W,
                                                   float* __restrict__ Out,
                                                   const float* __restrict__ res,
                                                   int M, int N, int K) {
    int lane = threadIdx.x & 31;
    int wave = threadIdx.x >> 5;
    int row0 = blockIdx.y * 128 + wave * 16;
    int col0 = blockIdx.x * 64;
    v8f acc[4];
#pragma unroll
    for (int t = 0; t < 4; ++t) acc[t] = {};
    for (int k0 = 0; k0 < K; k0 += 32) {
        v16bf a = frag_a(A, row0, k0, K, lane);
#pragma unroll
        for (int t = 0; t < 4; ++t) {
            v16bf b = frag_b_clamp(W, col0 + t * 16, k0, K, N, lane);
            acc[t] = wmma_bf16(a, b, acc[t]);
        }
    }
    int c = lane & 15, rb = (lane >> 4) * 8;
#pragma unroll
    for (int t = 0; t < 4; ++t) {
        int col = col0 + t * 16 + c;
        if (col < N) {
#pragma unroll
            for (int v = 0; v < 8; ++v) {
                size_t o = (size_t)(row0 + rb + v) * N + col;
                float val = acc[t][v];
                if (res) val += res[o];
                Out[o] = val;
            }
        }
    }
}

// K3: dt softplus, depthwise conv + SiLU; emit x_in, xdt^T, B, B^T, C
__global__ __launch_bounds__(256) void k_dtconv(const float* __restrict__ zx,
                                                const float* __restrict__ convw,
                                                const float* __restrict__ convb,
                                                const float* __restrict__ dtb,
                                                float* __restrict__ dtg,
                                                float* __restrict__ xin,
                                                bf16* __restrict__ xdtT,
                                                bf16* __restrict__ Bbf,
                                                bf16* __restrict__ Cbf,
                                                bf16* __restrict__ BTg) {
    int row = blockIdx.x;
    int b = row >> 11, l = row & (L_N - 1);
    int cch = l >> 5, s = l & 31;
    int tid = threadIdx.x;
    __shared__ float sdt[NHEADS];
    if (tid < NHEADS) {
        float v = zx[(size_t)row * D_IN_PROJ + (D_INNER + CONV_DIM) + tid] + dtb[tid];
        float sp = (v > 20.f) ? v : log1pf(expf(v));
        sdt[tid] = sp;
        dtg[(size_t)row * NHEADS + tid] = sp;
    }
    __syncthreads();
    for (int ch = tid; ch < CONV_DIM; ch += 256) {
        float acc = convb[ch];
#pragma unroll
        for (int k = 0; k < 4; ++k) {
            int t = l - 3 + k;
            if (t >= 0)
                acc += zx[(size_t)(b * L_N + t) * D_IN_PROJ + D_INNER + ch] *
                       convw[ch * 4 + k];
        }
        float sv = acc / (1.f + expf(-acc));
        if (ch < D_INNER) {
            int h = ch / HEADDIM, p = ch % HEADDIM;
            xin[(size_t)row * D_INNER + ch] = sv;
            size_t bch = (size_t)((b * NCHUNK + cch) * NHEADS + h);
            xdtT[(bch * 64 + p) * 32 + s] = f2bf(sv * sdt[h]);
        } else if (ch < D_INNER + D_STATE) {
            int n = ch - D_INNER;
            bf16 bv = f2bf(sv);
            Bbf[(size_t)row * D_STATE + n] = bv;
            BTg[((size_t)(b * NCHUNK + cch) * D_STATE + n) * 32 + s] = bv;
        } else {
            int n = ch - (D_INNER + D_STATE);
            Cbf[(size_t)row * D_STATE + n] = f2bf(sv);
        }
    }
}

// K4: per-(b,h,chunk) cumsum of dt*A via wave32 prefix; chunk decay; pad xdtT
__global__ __launch_bounds__(32) void k_cumsum(const float* __restrict__ dtg,
                                               const float* __restrict__ Alog,
                                               float* __restrict__ acum,
                                               float* __restrict__ cdg,
                                               bf16* __restrict__ xdtT) {
    int id = blockIdx.x;
    int h = id % NHEADS, cch = (id / NHEADS) % NCHUNK, b = id / (NHEADS * NCHUNK);
    int lane = threadIdx.x;
    float A = -expf(Alog[h]);
    int l = cch * CHUNK + lane;
    float v = dtg[(size_t)(b * L_N + l) * NHEADS + h] * A;
#pragma unroll
    for (int off = 1; off < 32; off <<= 1) {
        float t = __shfl_up(v, off, 32);
        if (lane >= off) v += t;
    }
    acum[(size_t)(b * NHEADS + h) * L_N + l] = v;
    if (lane == 31) cdg[(b * NHEADS + h) * NCHUNK + cch] = expf(v);
    size_t base = (size_t)((b * NCHUNK + cch) * NHEADS + h) * 64 * 32;
#pragma unroll
    for (int p = 56; p < 64; ++p) xdtT[base + p * 32 + lane] = f2bf(0.f);
}

// K5: G = C * B^T per (b,chunk): 32x32, K=128 (bf16 WMMA)
__global__ __launch_bounds__(32) void k_gmat(const bf16* __restrict__ Cbf,
                                             const bf16* __restrict__ Bbf,
                                             float* __restrict__ G) {
    int bc = blockIdx.x;                     // (b*NCHUNK + c)
    int lane = threadIdx.x;
    const bf16* Crow = Cbf + (size_t)bc * CHUNK * D_STATE;
    const bf16* Brow = Bbf + (size_t)bc * CHUNK * D_STATE;
    int cc = lane & 15, rb = (lane >> 4) * 8;
#pragma unroll
    for (int lt = 0; lt < 2; ++lt) {
        v8f acc[2]; acc[0] = {}; acc[1] = {};
        for (int k0 = 0; k0 < D_STATE; k0 += 32) {
            v16bf a = frag_a(Crow, lt * 16, k0, D_STATE, lane);
#pragma unroll
            for (int st = 0; st < 2; ++st) {
                v16bf bb = frag_b(Brow, st * 16, k0, D_STATE, lane);
                acc[st] = wmma_bf16(a, bb, acc[st]);
            }
        }
#pragma unroll
        for (int st = 0; st < 2; ++st)
#pragma unroll
            for (int v = 0; v < 8; ++v)
                G[(size_t)bc * 1024 + (lt * 16 + rb + v) * 32 + st * 16 + cc] = acc[st][v];
    }
}

// K6: Ydiag (masked L ⊙ G times xdt) and intra-chunk states, per (b,c,h)
__global__ __launch_bounds__(64) void k_ydiag_states(const float* __restrict__ G,
                                                     const float* __restrict__ acum,
                                                     const bf16* __restrict__ xdtT,
                                                     const bf16* __restrict__ BTg,
                                                     float* __restrict__ yraw,
                                                     float* __restrict__ stg) {
    int id = blockIdx.x;
    int h = id % NHEADS, cch = (id / NHEADS) % NCHUNK, b = id / (NHEADS * NCHUNK);
    int wave = threadIdx.x >> 5, lane = threadIdx.x & 31;
    const float* ac = acum + (size_t)(b * NHEADS + h) * L_N + cch * CHUNK;
    size_t bch = (size_t)((b * NCHUNK + cch) * NHEADS + h);
    const bf16* xT = xdtT + bch * 64 * 32;
    int r = lane & 15, hh = lane >> 4;
    int cc = lane & 15, rb = hh * 8;

    if (wave == 0) {  // Ydiag: [32 x 64pad] = Mh[32x32] * xdtT^T
        const float* Gp = G + (size_t)(b * NCHUNK + cch) * 1024;
#pragma unroll
        for (int mt = 0; mt < 2; ++mt) {
            int lrow = mt * 16 + r;
            float acl = ac[lrow];
            v16bf afr;
#pragma unroll
            for (int j = 0; j < 8; ++j) {
                int s1 = hh * 8 + j, s2 = 16 + hh * 8 + j;
                float g1 = (s1 <= lrow) ? Gp[lrow * 32 + s1] * expf(acl - ac[s1]) : 0.f;
                float g2 = (s2 <= lrow) ? Gp[lrow * 32 + s2] * expf(acl - ac[s2]) : 0.f;
                afr[j] = f2bf(g1); afr[j + 8] = f2bf(g2);
            }
#pragma unroll
            for (int pt = 0; pt < 4; ++pt) {
                v16bf bfr = frag_b(xT, pt * 16, 0, 32, lane);
                v8f acc = {};
                acc = wmma_bf16(afr, bfr, acc);
                int p = pt * 16 + cc;
                if (p < HEADDIM) {
#pragma unroll
                    for (int v = 0; v < 8; ++v) {
                        int lr = mt * 16 + rb + v;
                        yraw[(size_t)(b * L_N + cch * CHUNK + lr) * D_INNER +
                             h * HEADDIM + p] = acc[v];
                    }
                }
            }
        }
    }
    // states[p,n] = sum_l (dec[l]*xdt[l,p]) * B[l,n]; split n-tiles across waves
    const bf16* BT = BTg + (size_t)(b * NCHUNK + cch) * D_STATE * 32;
    float aclast = ac[31];
#pragma unroll
    for (int pt = 0; pt < 4; ++pt) {
        int p = pt * 16 + r;
        const bf16* xrow = xT + p * 32;
        v16bf afr;
#pragma unroll
        for (int j = 0; j < 8; ++j) {
            int k1 = hh * 8 + j, k2 = 16 + hh * 8 + j;
            afr[j]     = f2bf(bf2f(xrow[k1]) * expf(aclast - ac[k1]));
            afr[j + 8] = f2bf(bf2f(xrow[k2]) * expf(aclast - ac[k2]));
        }
#pragma unroll
        for (int i = 0; i < 4; ++i) {
            int nt = wave * 4 + i;
            v16bf bfr = frag_b(BT, nt * 16, 0, 32, lane);
            v8f acc = {};
            acc = wmma_bf16(afr, bfr, acc);
            int n = nt * 16 + cc;
#pragma unroll
            for (int v = 0; v < 8; ++v) {
                int pp = pt * 16 + rb + v;
                if (pp < HEADDIM)
                    stg[(bch * HEADDIM + pp) * D_STATE + n] = acc[v];
            }
        }
    }
}

// K7: in-place sequential inter-chunk scan: prev_c = carry; carry = carry*cd + st
__global__ __launch_bounds__(256) void k_scan(float* __restrict__ stg,
                                              const float* __restrict__ cdg) {
    int idx = blockIdx.x * 256 + threadIdx.x;   // B*NH*HD*DS = 688128 exactly
    int n = idx & (D_STATE - 1);
    int rest = idx >> 7;
    int p = rest % HEADDIM; rest /= HEADDIM;
    int h = rest % NHEADS;
    int b = rest / NHEADS;
    const float* cdp = cdg + (b * NHEADS + h) * NCHUNK;
    float carry = 0.f;
    for (int cch = 0; cch < NCHUNK; ++cch) {
        size_t o = (((size_t)((b * NCHUNK + cch) * NHEADS + h) * HEADDIM + p) *
                    D_STATE) + n;
        float tmp = stg[o];
        stg[o] = carry;
        carry = carry * cdp[cch] + tmp;
    }
}

// K8: Yoff += exp(Acum)[l] * C[l,:] · prev[h]^T, per (b,c,h)
__global__ __launch_bounds__(32) void k_yoff(const bf16* __restrict__ Cbf,
                                             const float* __restrict__ acum,
                                             const float* __restrict__ stg,
                                             float* __restrict__ yraw) {
    int id = blockIdx.x;
    int h = id % NHEADS, cch = (id / NHEADS) % NCHUNK, b = id / (NHEADS * NCHUNK);
    int lane = threadIdx.x;
    int r = lane & 15, hh = lane >> 4;
    int cc = lane & 15, rb = hh * 8;
    const float* ac = acum + (size_t)(b * NHEADS + h) * L_N + cch * CHUNK;
    const bf16* Crow = Cbf + (size_t)(b * L_N + cch * CHUNK) * D_STATE;
    const float* prev = stg + (size_t)((b * NCHUNK + cch) * NHEADS + h) *
                              HEADDIM * D_STATE;
#pragma unroll
    for (int mt = 0; mt < 2; ++mt) {
        int lrow = mt * 16 + r;
        float e1 = expf(ac[lrow]);
        v8f acc[4];
#pragma unroll
        for (int t = 0; t < 4; ++t) acc[t] = {};
        for (int k0 = 0; k0 < D_STATE; k0 += 32) {
            const bf16* crow = Crow + (size_t)lrow * D_STATE + k0 + hh * 8;
            v16bf afr;
#pragma unroll
            for (int j = 0; j < 8; ++j) {
                afr[j]     = f2bf(bf2f(crow[j]) * e1);
                afr[j + 8] = f2bf(bf2f(crow[16 + j]) * e1);
            }
#pragma unroll
            for (int pt = 0; pt < 4; ++pt) {
                int pc = pt * 16 + cc; pc = (pc < HEADDIM) ? pc : (HEADDIM - 1);
                const float* prow = prev + (size_t)pc * D_STATE + k0 + hh * 8;
                v16bf bfr;
#pragma unroll
                for (int j = 0; j < 8; ++j) { bfr[j] = f2bf(prow[j]); bfr[j + 8] = f2bf(prow[16 + j]); }
                acc[pt] = wmma_bf16(afr, bfr, acc[pt]);
            }
        }
#pragma unroll
        for (int pt = 0; pt < 4; ++pt) {
            int p = pt * 16 + cc;
            if (p < HEADDIM) {
#pragma unroll
                for (int v = 0; v < 8; ++v) {
                    int lr = mt * 16 + rb + v;
                    yraw[(size_t)(b * L_N + cch * CHUNK + lr) * D_INNER +
                         h * HEADDIM + p] += acc[pt][v];
                }
            }
        }
    }
}

// K9: y = rmsnorm((yraw + x_in*D) * silu(z)) * w  -> bf16
__global__ __launch_bounds__(256) void k_gatednorm(const float* __restrict__ yraw,
                                                   const float* __restrict__ zx,
                                                   const float* __restrict__ xin,
                                                   const float* __restrict__ Dp,
                                                   const float* __restrict__ gw,
                                                   bf16* __restrict__ ybf) {
    int row = blockIdx.x, tid = threadIdx.x;
    __shared__ float ts[D_INNER];
    __shared__ float red[256];
    float s = 0.f;
    for (int i = tid; i < D_INNER; i += 256) {
        float z = zx[(size_t)row * D_IN_PROJ + i];
        float sz = z / (1.f + expf(-z));
        int h = i / HEADDIM;
        float t = (yraw[(size_t)row * D_INNER + i] +
                   xin[(size_t)row * D_INNER + i] * Dp[h]) * sz;
        ts[i] = t;
        s += t * t;
    }
    red[tid] = s; __syncthreads();
    for (int st = 128; st > 0; st >>= 1) {
        if (tid < st) red[tid] += red[tid + st];
        __syncthreads();
    }
    float rms = rsqrtf(red[0] / (float)D_INNER + EPSF);
    for (int i = tid; i < D_INNER; i += 256)
        ybf[(size_t)row * D_INNER + i] = f2bf(ts[i] * rms * gw[i]);
}

// ---------------------------------------------------------------------------
extern "C" void kernel_launch(void* const* d_in, const int* in_sizes, int n_in,
                              void* d_out, int out_size, void* d_ws, size_t ws_size,
                              hipStream_t stream) {
    (void)in_sizes; (void)n_in; (void)out_size; (void)ws_size;
    const float* hs        = (const float*)d_in[0];
    const float* norm_w    = (const float*)d_in[1];
    const float* in_proj_w = (const float*)d_in[2];
    const float* conv_w    = (const float*)d_in[3];
    const float* conv_b    = (const float*)d_in[4];
    const float* dt_bias   = (const float*)d_in[5];
    const float* A_log     = (const float*)d_in[6];
    const float* Dv        = (const float*)d_in[7];
    const float* gnw       = (const float*)d_in[8];
    const float* out_proj_w= (const float*)d_in[9];
    float* out = (float*)d_out;

    // ---- carve workspace ----
    char* p = (char*)d_ws;
    auto take = [&](size_t bytes) {
        char* r = p;
        p += (bytes + 255) & ~(size_t)255;
        return (void*)r;
    };
    float* zx    = (float*)take((size_t)MROWS * D_IN_PROJ * 4);
    bf16*  xbf   = (bf16*) take((size_t)MROWS * D_MODEL * 2);
    bf16*  winb  = (bf16*) take((size_t)D_IN_PROJ * D_MODEL * 2);
    bf16*  woutb = (bf16*) take((size_t)D_MODEL * D_INNER * 2);
    float* dtg   = (float*)take((size_t)MROWS * NHEADS * 4);
    float* acum  = (float*)take((size_t)B_N * NHEADS * L_N * 4);
    float* cdg   = (float*)take((size_t)B_N * NHEADS * NCHUNK * 4);
    float* xin   = (float*)take((size_t)MROWS * D_INNER * 4);
    bf16*  xdtT  = (bf16*) take((size_t)B_N * NCHUNK * NHEADS * 64 * 32 * 2);
    bf16*  Bbf   = (bf16*) take((size_t)MROWS * D_STATE * 2);
    bf16*  Cbf   = (bf16*) take((size_t)MROWS * D_STATE * 2);
    bf16*  BTg   = (bf16*) take((size_t)B_N * NCHUNK * D_STATE * 32 * 2);
    float* Gg    = (float*)take((size_t)B_N * NCHUNK * 32 * 32 * 4);
    float* stg   = (float*)take((size_t)B_N * NCHUNK * NHEADS * HEADDIM * D_STATE * 4);
    float* yraw  = (float*)take((size_t)MROWS * D_INNER * 4);
    bf16*  ybf   = (bf16*) take((size_t)MROWS * D_INNER * 2);

    const int n1 = D_IN_PROJ * D_MODEL;   // 1,994,496
    const int n2 = D_MODEL * D_INNER;     //   903,168
    k_cvt<<<dim3((n1 + n2 + 255) / 256), dim3(256), 0, stream>>>(
        in_proj_w, n1, out_proj_w, n2, winb, woutb);

    k_rmsnorm<<<dim3(MROWS), dim3(256), 0, stream>>>(hs, norm_w, xbf);

    // in_proj: [8192x672] x [2968x672]^T -> zx
    k_gemm_bf16<<<dim3((D_IN_PROJ + 63) / 64, MROWS / 128), dim3(256), 0, stream>>>(
        xbf, winb, zx, nullptr, MROWS, D_IN_PROJ, D_MODEL);

    k_dtconv<<<dim3(MROWS), dim3(256), 0, stream>>>(
        zx, conv_w, conv_b, dt_bias, dtg, xin, xdtT, Bbf, Cbf, BTg);

    k_cumsum<<<dim3(B_N * NHEADS * NCHUNK), dim3(32), 0, stream>>>(
        dtg, A_log, acum, cdg, xdtT);

    k_gmat<<<dim3(B_N * NCHUNK), dim3(32), 0, stream>>>(Cbf, Bbf, Gg);

    k_ydiag_states<<<dim3(B_N * NCHUNK * NHEADS), dim3(64), 0, stream>>>(
        Gg, acum, xdtT, BTg, yraw, stg);

    k_scan<<<dim3((B_N * NHEADS * HEADDIM * D_STATE) / 256), dim3(256), 0, stream>>>(
        stg, cdg);

    k_yoff<<<dim3(B_N * NCHUNK * NHEADS), dim3(32), 0, stream>>>(
        Cbf, acum, stg, yraw);

    k_gatednorm<<<dim3(MROWS), dim3(256), 0, stream>>>(
        yraw, zx, xin, Dv, gnw, ybf);

    // out_proj + residual: [8192x1344] x [672x1344]^T + hs -> out
    k_gemm_bf16<<<dim3((D_MODEL + 63) / 64, MROWS / 128), dim3(256), 0, stream>>>(
        ybf, woutb, out, hs, MROWS, D_MODEL, D_INNER);
}
